// HelformerAttention_84636625535577
// MI455X (gfx1250) — matrix operand order repaired and checked by reading, not compiled, not run on hardware
//
#include <hip/hip_runtime.h>
#include <hip/hip_bf16.h>
#include <stddef.h>

// ---------------------------------------------------------------------------
// Helformer for MI455X (gfx1250): wave32 + v_wmma_f32_16x16x32_bf16 everywhere
// ---------------------------------------------------------------------------

typedef __bf16 bf16;
typedef __attribute__((ext_vector_type(8)))  float v8f;
typedef __attribute__((ext_vector_type(8)))  bf16  bf16x8;
typedef __attribute__((ext_vector_type(16))) bf16  bf16x16;

#define Bk  4
#define Sk  1024
#define Dk  1024
#define Hk  16
#define HDk 64
#define NSk 12
#define AP  (Dk + 16)   // padded bf16 A-row stride (avoid LDS bank conflicts)

// d_out layout (float element offsets), reference return order:
//   output [B,S,D], attn [B,H,S,S], level_f [B,D], trend_f [B,D], hw [B,S,D]
#define OUT_ATTN 4194304ULL
#define OUT_LVL  71303168ULL
#define OUT_TRD  71307264ULL
#define OUT_HW   71311360ULL

// ------------------------------- fragments ---------------------------------
__device__ __forceinline__ bf16x16 make_frag(const bf16* p0, const bf16* p1) {
  bf16x8 lo = *(const bf16x8*)p0;
  bf16x8 hi = *(const bf16x8*)p1;
  bf16x16 r;
#pragma unroll
  for (int i = 0; i < 8; ++i) { r[i] = lo[i]; r[i + 8] = hi[i]; }
  return r;
}
__device__ __forceinline__ v8f zero_acc() {
  v8f c;
#pragma unroll
  for (int i = 0; i < 8; ++i) c[i] = 0.0f;
  return c;
}
__device__ __forceinline__ v8f wmma_bf(bf16x16 a, bf16x16 b, v8f c) {
  return __builtin_amdgcn_wmma_f32_16x16x32_bf16(false, a, false, b,
                                                 (short)0, c, false, false);
}
// A fragment for 16-bit 16x32: lane<16: elems0-7=k0..k0+7, 8-15=k0+16..23;
// lane>=16: elems0-7=k0+8..15, 8-15=k0+24..31  (hi = lane>>4)
__device__ __forceinline__ bf16x16 a_frag(const bf16* rowbase, int k0, int hi) {
  const bf16* p = rowbase + k0 + hi * 8;
  return make_frag(p, p + 16);
}
// B fragment (KxN, 16-bit): same K order as A fragment (elems0-7=k0+8*hi..,
// 8-15=k0+16+8*hi..) of column n, weights kept in [n][k] row-major.
__device__ __forceinline__ bf16x16 b_frag(const bf16* colbase, int k0, int hi) {
  const bf16* p = colbase + k0 + hi * 8;
  return make_frag(p, p + 16);
}

// ---------------------------- weight conversion ----------------------------
__global__ void convert_f32_bf16(const float* __restrict__ src,
                                 bf16* __restrict__ dst, int n) {
  int i = blockIdx.x * blockDim.x + threadIdx.x;
  if (i < n) dst[i] = (bf16)src[i];
}

// ------------------------- sequential HW recurrence ------------------------
// One persistent workgroup (32 waves, one WGP). State lives in LDS.
// B=4 rows padded to M=16: lanes whose logical A-row >= 4 just load a
// duplicate of row (lane&3); D rows 4..15 are never read, so no masking or
// zero-fill is needed in the inner WMMA loop.
__global__ __launch_bounds__(1024) void helformer_recurrence(
    const float* __restrict__ x, const int* __restrict__ tidx,
    const float* __restrict__ alpha, const float* __restrict__ beta,
    const float* __restrict__ gamma, const float* __restrict__ seas0,
    const float* __restrict__ bl, const float* __restrict__ bt,
    const float* __restrict__ bs,
    const bf16* __restrict__ Wl, const bf16* __restrict__ Wt,
    const bf16* __restrict__ Ws, float* __restrict__ outbase) {
  __shared__ float s_level[Bk][Dk];
  __shared__ float s_trend[Bk][Dk];
  __shared__ float s_buf[NSk][Dk];
  __shared__ float s_lvlnew[Bk][Dk];
  __shared__ float s_su[Bk][Dk];
  __shared__ bf16  s_A0[Bk][AP];
  __shared__ bf16  s_A1[Bk][AP];
  __shared__ bf16  s_A2[Bk][AP];

  const int tid = threadIdx.x;
  const int lane = tid & 31;
  const int wv = tid >> 5;
  const int row = lane & 15;
  const int arow = lane & 3;   // duplicated A row for padded lanes
  const int hi = lane >> 4;

  const float sa = 1.0f / (1.0f + __expf(-alpha[0]));
  const float sb = 1.0f / (1.0f + __expf(-beta[0]));
  const float sg = 1.0f / (1.0f + __expf(-gamma[0]));

  float* lvl = &s_level[0][0];
  float* trd = &s_trend[0][0];
  float* bufp = &s_buf[0][0];
  for (int i = tid; i < Bk * Dk; i += 1024) { lvl[i] = 0.0f; trd[i] = 0.0f; }
  for (int i = tid; i < NSk * Dk; i += 1024) bufp[i] = seas0[i];
  __syncthreads();

  float* hw = outbase + OUT_HW;
  float* lvlF = outbase + OUT_LVL;
  float* trdF = outbase + OUT_TRD;

  for (int t = 0; t < Sk; ++t) {
    const int ssum = tidx[t] + tidx[Sk + t] + tidx[2 * Sk + t] + tidx[3 * Sk + t];
    const int idx = (ssum / Bk) % NSk;

    // phase 0: build A for level GEMM
#pragma unroll
    for (int e = 0; e < 4; ++e) {
      int linear = tid + e * 1024;
      int m = linear >> 10, n = linear & (Dk - 1);
      float xv = x[(size_t)(m * Sk + t) * Dk + n];
      float a = sa * (xv - s_buf[idx][n]) +
                (1.0f - sa) * (s_level[m][n] + s_trend[m][n]);
      s_A0[m][n] = (bf16)a;
    }
    __syncthreads();

    // phase 1: level GEMM (each wave: 2 N-tiles of 16)
    {
      v8f c0 = zero_acc(), c1 = zero_acc();
      const int n0 = (wv * 2) * 16 + row;
      const int n1 = (wv * 2 + 1) * 16 + row;
      const bf16* b0 = Wl + (size_t)n0 * Dk;
      const bf16* b1 = Wl + (size_t)n1 * Dk;
      const bf16* a0 = &s_A0[arow][0];
#pragma unroll 4
      for (int k0 = 0; k0 < Dk; k0 += 32) {
        bf16x16 af = a_frag(a0, k0, hi);
        __builtin_prefetch(b0 + k0 + 512, 0, 3);
        c0 = wmma_bf(af, b_frag(b0, k0, hi), c0);
        c1 = wmma_bf(af, b_frag(b1, k0, hi), c1);
      }
      if (lane < 16) {
        int nn0 = (wv * 2) * 16 + lane;
        int nn1 = (wv * 2 + 1) * 16 + lane;
#pragma unroll
        for (int m = 0; m < Bk; ++m) {
          s_lvlnew[m][nn0] = c0[m] + bl[nn0];
          s_lvlnew[m][nn1] = c1[m] + bl[nn1];
        }
      }
    }
    __syncthreads();

    // phase 2: build A for trend + seasonal GEMMs; commit level := level_new
#pragma unroll
    for (int e = 0; e < 4; ++e) {
      int linear = tid + e * 1024;
      int m = linear >> 10, n = linear & (Dk - 1);
      float lvn = s_lvlnew[m][n];
      float lo = s_level[m][n];
      float to = s_trend[m][n];
      float xv = x[(size_t)(m * Sk + t) * Dk + n];
      s_A1[m][n] = (bf16)(sb * (lvn - lo) + (1.0f - sb) * to);
      s_A2[m][n] = (bf16)(sg * (xv - lvn) + (1.0f - sg) * s_buf[idx][n]);
      s_level[m][n] = lvn;
    }
    __syncthreads();

    // phase 3: trend GEMM (waves 0-15) and seasonal GEMM (waves 16-31)
    {
      const bool is_tr = (wv < 16);
      const bf16* Amat = is_tr ? &s_A1[arow][0] : &s_A2[arow][0];
      const bf16* Wm = is_tr ? Wt : Ws;
      const float* bias = is_tr ? bt : bs;
      float(*dst)[Dk] = is_tr ? s_trend : s_su;
      const int wq = is_tr ? wv : (wv - 16);
      v8f c[4];
#pragma unroll
      for (int j = 0; j < 4; ++j) c[j] = zero_acc();
#pragma unroll 2
      for (int k0 = 0; k0 < Dk; k0 += 32) {
        bf16x16 af = a_frag(Amat, k0, hi);
        __builtin_prefetch(Wm + (size_t)(wq * 64 + row) * Dk + k0 + 512, 0, 3);
#pragma unroll
        for (int j = 0; j < 4; ++j) {
          int n = (wq * 4 + j) * 16 + row;
          c[j] = wmma_bf(af, b_frag(Wm + (size_t)n * Dk, k0, hi), c[j]);
        }
      }
      if (lane < 16) {
#pragma unroll
        for (int j = 0; j < 4; ++j) {
          int nn = (wq * 4 + j) * 16 + lane;
#pragma unroll
          for (int m = 0; m < Bk; ++m) dst[m][nn] = c[j][m] + bias[nn];
        }
      }
    }
    __syncthreads();

    // phase 4: emit hw, update seasonal buffer (batch mean), final carries
#pragma unroll
    for (int e = 0; e < 4; ++e) {
      int linear = tid + e * 1024;
      int m = linear >> 10, n = linear & (Dk - 1);
      hw[(size_t)(m * Sk + t) * Dk + n] =
          s_lvlnew[m][n] + s_trend[m][n] + s_su[m][n];
      if (e == 0)
        s_buf[idx][n] =
            0.25f * (s_su[0][n] + s_su[1][n] + s_su[2][n] + s_su[3][n]);
      if (t == Sk - 1) {
        lvlF[m * Dk + n] = s_lvlnew[m][n];
        trdF[m * Dk + n] = s_trend[m][n];
      }
    }
    __syncthreads();
  }
}

// ------------------------- fused QKV projections ---------------------------
// enhanced = hw + temporal_emb[t%NS]; Q,K bf16 [B,H,S,HD]; V transposed
// bf16 [B,H,HD,S]. Block: 64x64 output tile, A staged once for 3 weights.
__global__ __launch_bounds__(256) void qkv_proj(
    const float* __restrict__ hw, const int* __restrict__ tidx,
    const float* __restrict__ emb, const bf16* __restrict__ Wq,
    const bf16* __restrict__ Wk, const bf16* __restrict__ Wv,
    const float* __restrict__ bq, const float* __restrict__ bk,
    const float* __restrict__ bv, bf16* __restrict__ Qo,
    bf16* __restrict__ Ko, bf16* __restrict__ VTo) {
  __shared__ bf16 s_a[64][40];
  __shared__ int s_tm[64];
  const int tid = threadIdx.x, lane = tid & 31, wv = tid >> 5;
  const int bm = blockIdx.x;  // 64-row M block
  const int bn = blockIdx.y;  // 64-col N block (== head)
  const int mt = wv >> 1;
  const int ntp = (wv & 1) * 2;
  const int row = lane & 15, hi = lane >> 4;

  if (tid < 64) {
    int rg = bm * 64 + tid;
    s_tm[tid] = tidx[(rg >> 10) * Sk + (rg & (Sk - 1))] % NSk;
  }

  v8f cq0 = zero_acc(), cq1 = zero_acc();
  v8f ck0 = zero_acc(), ck1 = zero_acc();
  v8f cv0 = zero_acc(), cv1 = zero_acc();

  const int n0 = bn * 64 + ntp * 16 + row;
  const int n1 = n0 + 16;

  for (int kc = 0; kc < Dk; kc += 32) {
    __syncthreads();
#pragma unroll
    for (int e = 0; e < 8; ++e) {
      int linear = tid + e * 256;
      int r = linear >> 5, kk = linear & 31;
      int rg = bm * 64 + r;
      float v = hw[(size_t)rg * Dk + kc + kk] +
                emb[(size_t)s_tm[r] * Dk + kc + kk];
      s_a[r][kk] = (bf16)v;
    }
    __syncthreads();
    bf16x16 af = a_frag(&s_a[mt * 16 + row][0], 0, hi);
    size_t o0 = (size_t)n0 * Dk + kc;
    size_t o1 = (size_t)n1 * Dk + kc;
    __builtin_prefetch(Wq + o0 + 512, 0, 3);
    cq0 = wmma_bf(af, b_frag(Wq + o0, 0, hi), cq0);
    cq1 = wmma_bf(af, b_frag(Wq + o1, 0, hi), cq1);
    ck0 = wmma_bf(af, b_frag(Wk + o0, 0, hi), ck0);
    ck1 = wmma_bf(af, b_frag(Wk + o1, 0, hi), ck1);
    cv0 = wmma_bf(af, b_frag(Wv + o0, 0, hi), cv0);
    cv1 = wmma_bf(af, b_frag(Wv + o1, 0, hi), cv1);
  }

  const int c0 = bn * 64 + ntp * 16 + (lane & 15);
  const int c1 = c0 + 16;
  const int h0 = c0 >> 6, hd0 = c0 & 63;
  const int h1 = c1 >> 6, hd1 = c1 & 63;
#pragma unroll
  for (int v = 0; v < 8; ++v) {
    int rg = bm * 64 + mt * 16 + v + hi * 8;
    int b = rg >> 10, s = rg & (Sk - 1);
    size_t q0 = ((size_t)(b * Hk + h0) * Sk + s) * HDk + hd0;
    size_t q1 = ((size_t)(b * Hk + h1) * Sk + s) * HDk + hd1;
    Qo[q0] = (bf16)(cq0[v] + bq[c0]);
    Qo[q1] = (bf16)(cq1[v] + bq[c1]);
    Ko[q0] = (bf16)(ck0[v] + bk[c0]);
    Ko[q1] = (bf16)(ck1[v] + bk[c1]);
    VTo[((size_t)(b * Hk + h0) * HDk + hd0) * Sk + s] = (bf16)(cv0[v] + bv[c0]);
    VTo[((size_t)(b * Hk + h1) * HDk + hd1) * Sk + s] = (bf16)(cv1[v] + bv[c1]);
  }
}

// ------------------- fused scores + softmax + context ----------------------
// One block per (b,h,16 query rows): 16 waves. Scores in LDS, per-row wave
// softmax (attn written f32 to d_out), then attn(bf16) x V via WMMA.
__global__ __launch_bounds__(512) void attention(
    const bf16* __restrict__ Q, const bf16* __restrict__ K,
    const bf16* __restrict__ VT, float* __restrict__ attn_out,
    bf16* __restrict__ ctx) {
  __shared__ float s_sc[16][1032];
  __shared__ bf16 s_p[16][1040];
  const int tid = threadIdx.x, lane = tid & 31, wv = tid >> 5;
  const int blk = blockIdx.x;
  const int bh = blk >> 6;    // b*H + h
  const int qb = blk & 63;    // 16-row query block
  const int b = bh >> 4, h = bh & 15;
  const int row = lane & 15, hi = lane >> 4;

  // scores: each wave computes 4 N-tiles (keys)
  {
    const bf16* arow = Q + ((size_t)bh * Sk + qb * 16 + row) * HDk;
    v8f c[4];
#pragma unroll
    for (int j = 0; j < 4; ++j) c[j] = zero_acc();
#pragma unroll
    for (int k0 = 0; k0 < HDk; k0 += 32) {
      bf16x16 af = a_frag(arow, k0, hi);
#pragma unroll
      for (int j = 0; j < 4; ++j) {
        int n = (wv * 4 + j) * 16 + row;
        c[j] = wmma_bf(af, b_frag(K + ((size_t)bh * Sk + n) * HDk, k0, hi), c[j]);
      }
    }
#pragma unroll
    for (int j = 0; j < 4; ++j)
#pragma unroll
      for (int v = 0; v < 8; ++v)
        s_sc[v + hi * 8][(wv * 4 + j) * 16 + (lane & 15)] = c[j][v] * 0.125f;
  }
  __syncthreads();

  // softmax: wave wv owns query row wv
  {
    const int r = wv;
    float vals[32];
    float mx = -3.4e38f;
#pragma unroll
    for (int i = 0; i < 32; ++i) {
      float v = s_sc[r][lane + i * 32];
      vals[i] = v;
      mx = fmaxf(mx, v);
    }
#pragma unroll
    for (int o = 16; o > 0; o >>= 1) mx = fmaxf(mx, __shfl_xor(mx, o, 32));
    float sum = 0.0f;
#pragma unroll
    for (int i = 0; i < 32; ++i) {
      float e = __expf(vals[i] - mx);
      vals[i] = e;
      sum += e;
    }
#pragma unroll
    for (int o = 16; o > 0; o >>= 1) sum += __shfl_xor(sum, o, 32);
    const float inv = 1.0f / sum;
    const size_t abase = ((size_t)bh * Sk + qb * 16 + r) * Sk;
#pragma unroll
    for (int i = 0; i < 32; ++i) {
      float p = vals[i] * inv;
      attn_out[abase + lane + i * 32] = p;
      s_p[r][lane + i * 32] = (bf16)p;
    }
  }
  __syncthreads();

  // context: attn [16x1024] x V [1024x64]; waves 0..3 take one N-tile each
  if (wv < 4) {
    v8f c = zero_acc();
    const bf16* arow = &s_p[row][0];
    const int n = wv * 16 + row;
    const bf16* vcol = VT + ((size_t)bh * HDk + n) * Sk;
#pragma unroll 4
    for (int k0 = 0; k0 < Sk; k0 += 32)
      c = wmma_bf(a_frag(arow, k0, hi), b_frag(vcol, k0, hi), c);
#pragma unroll
    for (int v = 0; v < 8; ++v) {
      int s = qb * 16 + v + hi * 8;
      int d = h * HDk + wv * 16 + (lane & 15);
      ctx[((size_t)b * Sk + s) * Dk + d] = (bf16)c[v];
    }
  }
}

// --------------------------- output projection -----------------------------
__global__ __launch_bounds__(256) void out_proj(
    const bf16* __restrict__ ctx, const bf16* __restrict__ Wo,
    const float* __restrict__ bo, float* __restrict__ out) {
  const int tid = threadIdx.x, lane = tid & 31, wv = tid >> 5;
  const int bm = blockIdx.x, bn = blockIdx.y;
  const int mt = wv >> 1, ntp = (wv & 1) * 2;
  const int row = lane & 15, hi = lane >> 4;
  v8f c0 = zero_acc(), c1 = zero_acc();
  const bf16* arow = ctx + (size_t)(bm * 64 + mt * 16 + row) * Dk;
  const int n0 = bn * 64 + ntp * 16 + row;
  const bf16* b0 = Wo + (size_t)n0 * Dk;
  const bf16* b1 = Wo + (size_t)(n0 + 16) * Dk;
#pragma unroll 4
  for (int k0 = 0; k0 < Dk; k0 += 32) {
    bf16x16 af = a_frag(arow, k0, hi);
    __builtin_prefetch(b0 + k0 + 512, 0, 3);
    c0 = wmma_bf(af, b_frag(b0, k0, hi), c0);
    c1 = wmma_bf(af, b_frag(b1, k0, hi), c1);
  }
#pragma unroll
  for (int v = 0; v < 8; ++v) {
    int r = bm * 64 + mt * 16 + v + hi * 8;
    int cc0 = bn * 64 + ntp * 16 + (lane & 15);
    out[(size_t)r * Dk + cc0] = c0[v] + bo[cc0];
    out[(size_t)r * Dk + cc0 + 16] = c1[v] + bo[cc0 + 16];
  }
}

// -------------------------------- launcher ---------------------------------
extern "C" void kernel_launch(void* const* d_in, const int* in_sizes, int n_in,
                              void* d_out, int out_size, void* d_ws,
                              size_t ws_size, hipStream_t stream) {
  const float* x = (const float*)d_in[0];
  const int* tidx = (const int*)d_in[1];
  const float* alpha = (const float*)d_in[2];
  const float* beta = (const float*)d_in[3];
  const float* gamma = (const float*)d_in[4];
  const float* seas0 = (const float*)d_in[5];
  const float* Wl = (const float*)d_in[6];
  const float* bl = (const float*)d_in[7];
  const float* Wt = (const float*)d_in[8];
  const float* bt = (const float*)d_in[9];
  const float* Ws = (const float*)d_in[10];
  const float* bs = (const float*)d_in[11];
  const float* Wq = (const float*)d_in[12];
  const float* bq = (const float*)d_in[13];
  const float* Wk = (const float*)d_in[14];
  const float* bk = (const float*)d_in[15];
  const float* Wv = (const float*)d_in[16];
  const float* bv = (const float*)d_in[17];
  const float* Wo = (const float*)d_in[18];
  const float* bo = (const float*)d_in[19];
  const float* emb = (const float*)d_in[20];

  float* outp = (float*)d_out;
  char* ws = (char*)d_ws;

  const size_t MB = 1ull << 20;
  bf16* Wl_b = (bf16*)(ws + 0 * MB);
  bf16* Wt_b = (bf16*)(ws + 2 * MB);
  bf16* Ws_b = (bf16*)(ws + 4 * MB);
  bf16* Wq_b = (bf16*)(ws + 6 * MB);
  bf16* Wk_b = (bf16*)(ws + 8 * MB);
  bf16* Wv_b = (bf16*)(ws + 10 * MB);
  bf16* Wo_b = (bf16*)(ws + 12 * MB);
  bf16* Qb = (bf16*)(ws + 14 * MB);
  bf16* Kb = (bf16*)(ws + 22 * MB);
  bf16* VTb = (bf16*)(ws + 30 * MB);
  bf16* Ctx = (bf16*)(ws + 38 * MB);

  const int NW = Dk * Dk;
  dim3 cg(NW / 256);
  convert_f32_bf16<<<cg, 256, 0, stream>>>(Wl, Wl_b, NW);
  convert_f32_bf16<<<cg, 256, 0, stream>>>(Wt, Wt_b, NW);
  convert_f32_bf16<<<cg, 256, 0, stream>>>(Ws, Ws_b, NW);
  convert_f32_bf16<<<cg, 256, 0, stream>>>(Wq, Wq_b, NW);
  convert_f32_bf16<<<cg, 256, 0, stream>>>(Wk, Wk_b, NW);
  convert_f32_bf16<<<cg, 256, 0, stream>>>(Wv, Wv_b, NW);
  convert_f32_bf16<<<cg, 256, 0, stream>>>(Wo, Wo_b, NW);

  helformer_recurrence<<<1, 1024, 0, stream>>>(
      x, tidx, alpha, beta, gamma, seas0, bl, bt, bs, Wl_b, Wt_b, Ws_b, outp);

  qkv_proj<<<dim3(64, 16), 256, 0, stream>>>(outp + OUT_HW, tidx, emb, Wq_b,
                                             Wk_b, Wv_b, bq, bk, bv, Qb, Kb,
                                             VTb);

  attention<<<4096, 512, 0, stream>>>(Qb, Kb, VTb, outp + OUT_ATTN, Ctx);

  out_proj<<<dim3(64, 16), 256, 0, stream>>>(Ctx, Wo_b, bo, outp);
}
